// CVT_56590489092836
// MI455X (gfx1250) — matrix operand order repaired and checked by reading, not compiled, not run on hardware
//
#include <hip/hip_runtime.h>
#include <hip/hip_bf16.h>

// ---------------------------------------------------------------------------
// CDNA5 (gfx1250) wave32 WMMA types and helpers
// ---------------------------------------------------------------------------
typedef __attribute__((ext_vector_type(16))) _Float16 v16h;
typedef __attribute__((ext_vector_type(8)))  float    v8f;
typedef __attribute__((ext_vector_type(4)))  _Float16 h4;   // 8B packed halves
typedef float f4u __attribute__((ext_vector_type(4), aligned(4))); // 4B-aligned float4

#define WMMA_F16(a, b, c) \
    __builtin_amdgcn_wmma_f32_16x16x32_f16(false, (a), false, (b), (short)0, (c), false, false)

// Problem constants (match reference)
#define BB   8
#define CC   64
#define HH   64
#define WW   64
#define NN   4096   // H*W
#define QKC  8

// CDNA5 f16 fragment swizzle (wave32, 16x16x32):
//  A (16x32, [m][k]): lane = (m&15) + 16*((k>>3)&1), idx = ((k>>4)<<3) + (k&7)
//  B (32x16, [k][n]): lane = (n&15) + 16*((k>>3)&1), idx = ((k>>4)<<3) + (k&7)
//  C (16x16 f32):     m = r + 8*(lane>>4), n = lane&15
// LDS tiles are stored directly in fragment order: [tile][lane][16 halves],
// so a fragment load is one aligned 32-byte v16h read (2x ds_load_b128).

// ---------------------------------------------------------------------------
// Kernel 1: out[M,Ntot] = relu(X[M,K] @ W^T + bias), W:[Ntot,K] (PyTorch Linear)
// Workgroup tile: 64(M) x 128(N); 8 waves = 4(M) x 2(N); each wave 16x64.
// ---------------------------------------------------------------------------
__global__ void __launch_bounds__(256)
linear_relu_wmma(const float* __restrict__ X, const float* __restrict__ Wt,
                 const float* __restrict__ bias, float* __restrict__ out,
                 int M, int Ntot, int K)
{
    const int tid  = threadIdx.x;
    const int lane = tid & 31;
    const int wave = tid >> 5;
    const int wm   = wave & 3;
    const int wn   = wave >> 2;
    const int m0   = blockIdx.y * 64;
    const int n0   = blockIdx.x * 128;

    __shared__ v16h aS[4 * 32];   // A tile, fragment order, 4 m-tiles
    __shared__ v16h bS[8 * 32];   // B tile, fragment order, 8 n-tiles
    _Float16* aP = (_Float16*)aS;
    _Float16* bP = (_Float16*)bS;

    // Per-thread staging pointers, hoisted out of the K loop.
    // A: 64 rows x 8 quads = 512 quads -> 2 per thread.
    const float* xSrc[2]; _Float16* aDst[2];
#pragma unroll
    for (int u = 0; u < 2; ++u) {
        int i = tid + u * 256;
        int m = i >> 3, kq = (i & 7) << 2;
        xSrc[u] = X + (size_t)(m0 + m) * K + kq;
        int hi = (kq >> 3) & 1;
        int r2 = ((kq >> 4) << 3) + (kq & 7);            // quad-contiguous idx base
        aDst[u] = aP + ((((m >> 4) << 5) + (m & 15) + (hi << 4)) << 4) + r2;
    }
    // B: 128 rows x 8 quads = 1024 quads -> 4 per thread.
    const float* wSrc[4]; _Float16* bDst[4];
#pragma unroll
    for (int u = 0; u < 4; ++u) {
        int i = tid + u * 256;
        int n = i >> 3, kq = (i & 7) << 2;
        wSrc[u] = Wt + (size_t)(n0 + n) * K + kq;
        bDst[u] = bP + ((((n >> 4) << 5) + (n & 15) + (((kq >> 3) & 1) << 4)) << 4) + ((kq >> 4) << 3) + (kq & 7);
    }

    v8f acc[4];
#pragma unroll
    for (int t = 0; t < 4; ++t)
#pragma unroll
        for (int e = 0; e < 8; ++e) acc[t][e] = 0.0f;

    for (int k0 = 0; k0 < K; k0 += 32) {
        __syncthreads();
#pragma unroll
        for (int u = 0; u < 2; ++u) {
            f4u q = *(const f4u*)(xSrc[u] + k0);
            h4 h; h.x = (_Float16)q.x; h.y = (_Float16)q.y;
                  h.z = (_Float16)q.z; h.w = (_Float16)q.w;
            *(h4*)aDst[u] = h;
        }
#pragma unroll
        for (int u = 0; u < 4; ++u) {
            f4u q = *(const f4u*)(wSrc[u] + k0);
            h4 h; h.x = (_Float16)q.x; h.y = (_Float16)q.y;
                  h.z = (_Float16)q.z; h.w = (_Float16)q.w;
            *(h4*)bDst[u] = h;
        }
        if (k0 + 32 < K) {                 // gfx1250 global_prefetch_b8
            __builtin_prefetch(xSrc[0] + k0 + 32, 0, 1);
            __builtin_prefetch(wSrc[0] + k0 + 32, 0, 1);
        }
        __syncthreads();

        v16h a = aS[(wm << 5) + lane];     // 2x ds_load_b128
#pragma unroll
        for (int t = 0; t < 4; ++t) {
            v16h b = bS[(((wn << 2) + t) << 5) + lane];
            acc[t] = WMMA_F16(a, b, acc[t]);
        }
    }

    // Epilogue: bias + relu
#pragma unroll
    for (int t = 0; t < 4; ++t) {
        int n = n0 + wn * 64 + t * 16 + (lane & 15);
        float bv = bias[n];
#pragma unroll
        for (int r = 0; r < 8; ++r) {
            int m = m0 + wm * 16 + r + ((lane >> 4) << 3);
            float v = acc[t][r] + bv;
            out[(size_t)m * Ntot + n] = fmaxf(v, 0.0f);
        }
    }
}

// ---------------------------------------------------------------------------
// Kernel 2: implicit-GEMM conv (KSxKS, CIN in-channels, Cout<=64 out-channels)
// Tile: M=64 out-ch x N=128 pixels (two full image rows, W=64). 8 waves = 4m x 2n.
// MODE: 0 = relu, 1 = linear, 2 = fused final (second 64 in-channels are the
//       gathered value tensor v[.,.,idx]; epilogue out = addb + (conv+bias)*maxv)
// ---------------------------------------------------------------------------
template <int KS, int CIN, int MODE>
__global__ void __launch_bounds__(256)
conv_wmma(const float* __restrict__ in0,   // [B][64][N]
          const float* __restrict__ in1,   // fused: v [B][64][N]; else unused
          const float* __restrict__ wgt,   // [Cout][CIN][KS][KS]
          const float* __restrict__ bias,  // [Cout]
          float* __restrict__ out,         // [B][Cout][N]
          const int*   __restrict__ gidx,  // fused: [B][N]
          const float* __restrict__ maxv,  // fused: [B][N]
          const float* __restrict__ addb,  // fused: residual f_feat [B][64][N]
          int Cout)
{
    const int tid  = threadIdx.x;
    const int lane = tid & 31;
    const int wave = tid >> 5;
    const int wm   = wave & 3;
    const int wn   = wave >> 2;
    const int b    = blockIdx.y;
    const int p0   = blockIdx.x * 128;     // rows y0, y0+1
    const int bC   = b * CC;
    const int KS2  = KS * KS;

    __shared__ v16h aS[4 * 32];            // weights, fragment order
    __shared__ v16h bS[8 * 32];            // input slab, fragment order
    _Float16* aP = (_Float16*)aS;
    _Float16* bP = (_Float16*)bS;

    // Weights: 64 x 32 elements per chunk -> 8 scalars/thread.
    // Chunk source offset is wBase[u][ci0*KS2 + tap]; dst is fixed per thread.
    const float* wBase[8]; _Float16* aDst[8];
#pragma unroll
    for (int u = 0; u < 8; ++u) {
        int i = tid + u * 256;
        int o = i >> 5, kk = i & 31;
        wBase[u] = (o < Cout) ? (wgt + (size_t)(o * CIN + kk) * KS2) : nullptr;
        int lane0 = (o & 15) + (((kk >> 3) & 1) << 4);
        int idx   = ((kk >> 4) << 3) + (kk & 7);
        aDst[u] = aP + ((((o >> 4) << 5) + lane0) << 4) + idx;
    }
    // Input slab: 32 k x 128 pixels -> 4 quads/thread.
    int kkv[4], pqv[4]; _Float16* bDst[4];
#pragma unroll
    for (int u = 0; u < 4; ++u) {
        int i = tid + u * 256;
        int kk = i >> 5, pq = (i & 31) << 2;
        kkv[u] = kk; pqv[u] = pq;
        int lane0 = (pq & 15) + (((kk >> 3) & 1) << 4);
        bDst[u] = bP + ((((pq >> 4) << 5) + lane0) << 4) + ((kk >> 4) << 3) + (kk & 7);
    }

    v8f acc[4];
#pragma unroll
    for (int t = 0; t < 4; ++t)
#pragma unroll
        for (int e = 0; e < 8; ++e) acc[t][e] = 0.0f;

    for (int tap = 0; tap < KS2; ++tap) {
        const int kh = tap / KS, kw = tap % KS;
        const int dy = kh - KS / 2, dx = kw - KS / 2;
        for (int ci0 = 0; ci0 < CIN; ci0 += 32) {
            const int  g      = ci0 * KS2 + tap;
            const bool gather = (MODE == 2) && (ci0 >= 64);   // uniform per chunk

            __syncthreads();
            // Stage weights (zero rows o >= Cout)
#pragma unroll
            for (int u = 0; u < 8; ++u) {
                _Float16 h = (_Float16)0.0f;
                if (wBase[u]) h = (_Float16)wBase[u][g];
                *aDst[u] = h;
            }
            // Stage shifted / zero-padded / gathered input slab
#pragma unroll
            for (int u = 0; u < 4; ++u) {
                const int kk = kkv[u], pq = pqv[u];
                const int gp = p0 + pq;
                const int y  = gp >> 6, x0 = gp & 63;
                const int yy = y + dy,  xx0 = x0 + dx;
                _Float16* d = bDst[u];
                if ((unsigned)yy < (unsigned)HH) {
                    const int cc = (gather ? (ci0 - 64) : ci0) + kk;
                    const float* plane =
                        (gather ? in1 : in0) + (size_t)(bC + cc) * NN;
                    const int rowoff = yy * WW;
                    if (!gather && xx0 >= 0 && xx0 + 3 < WW) {
                        f4u q = *(const f4u*)(plane + rowoff + xx0);
                        d[0]  = (_Float16)q.x;
                        d[16] = (_Float16)q.y;
                        d[32] = (_Float16)q.z;
                        d[48] = (_Float16)q.w;
                    } else {
#pragma unroll
                        for (int j = 0; j < 4; ++j) {
                            int xx = xx0 + j;
                            float fv = 0.0f;
                            if ((unsigned)xx < (unsigned)WW) {
                                int pix = rowoff + xx;
                                fv = gather ? plane[gidx[b * NN + pix]]
                                            : plane[pix];
                            }
                            d[j * 16] = (_Float16)fv;
                        }
                    }
                } else {
                    d[0] = d[16] = d[32] = d[48] = (_Float16)0.0f;
                }
            }
            __syncthreads();

            v16h a = aS[(wm << 5) + lane];
#pragma unroll
            for (int t = 0; t < 4; ++t) {
                v16h bf = bS[(((wn << 2) + t) << 5) + lane];
                acc[t] = WMMA_F16(a, bf, acc[t]);
            }
        }
    }

    // Epilogue
#pragma unroll
    for (int t = 0; t < 4; ++t) {
        int p = p0 + wn * 64 + t * 16 + (lane & 15);
#pragma unroll
        for (int r = 0; r < 8; ++r) {
            int o = wm * 16 + r + ((lane >> 4) << 3);
            if (o < Cout) {
                float v  = acc[t][r] + bias[o];
                size_t oi = ((size_t)b * Cout + o) * NN + p;
                if (MODE == 0) {
                    out[oi] = fmaxf(v, 0.0f);
                } else if (MODE == 1) {
                    out[oi] = v;
                } else {
                    out[oi] = addb[((size_t)bC + o) * NN + p] + v * maxv[b * NN + p];
                }
            }
        }
    }
}

// ---------------------------------------------------------------------------
// Kernel 3: per-row streaming argmax of scores[b,i,j] = <k_i, q_j>, K = 8.
// One thread per output row i; q tiles staged in LDS. The [B,N,N] score
// tensor (512 MB) is never materialized.
// ---------------------------------------------------------------------------
__global__ void __launch_bounds__(256)
argmax_rows(const float* __restrict__ kmat, const float* __restrict__ qmat,
            int* __restrict__ idx, float* __restrict__ maxv)
{
    const int b = blockIdx.y;
    const int i = blockIdx.x * 256 + threadIdx.x;

    float kv[QKC];
#pragma unroll
    for (int c = 0; c < QKC; ++c)
        kv[c] = kmat[((size_t)b * QKC + c) * NN + i];

    __shared__ float qs[QKC][256];
    float best = -3.402823466e+38f;
    int   bi   = 0;

    for (int j0 = 0; j0 < NN; j0 += 256) {
        __syncthreads();
#pragma unroll
        for (int c = 0; c < QKC; ++c)
            qs[c][threadIdx.x] = qmat[((size_t)b * QKC + c) * NN + j0 + threadIdx.x];
        __syncthreads();
        for (int j = 0; j < 256; ++j) {
            float s = 0.0f;
#pragma unroll
            for (int c = 0; c < QKC; ++c) s = fmaf(kv[c], qs[c][j], s);
            if (s > best) { best = s; bi = j0 + j; }   // strict >: first-max semantics
        }
    }
    idx[b * NN + i]  = bi;
    maxv[b * NN + i] = best;
}

// ---------------------------------------------------------------------------
// Host-side orchestration
// ---------------------------------------------------------------------------
extern "C" void kernel_launch(void* const* d_in, const int* in_sizes, int n_in,
                              void* d_out, int out_size, void* d_ws, size_t ws_size,
                              hipStream_t stream)
{
    const float* img_feat = (const float*)d_in[0];
    const float* st_w1 = (const float*)d_in[1];
    const float* st_b1 = (const float*)d_in[2];
    const float* st_w2 = (const float*)d_in[3];
    const float* st_b2 = (const float*)d_in[4];
    const float* ft_w1 = (const float*)d_in[5];
    const float* ft_b1 = (const float*)d_in[6];
    const float* ft_w2 = (const float*)d_in[7];
    const float* ft_b2 = (const float*)d_in[8];
    const float* bt_w1 = (const float*)d_in[9];
    const float* bt_b1 = (const float*)d_in[10];
    const float* bt_w2 = (const float*)d_in[11];
    const float* bt_b2 = (const float*)d_in[12];
    const float* q_w   = (const float*)d_in[13];
    const float* q_b   = (const float*)d_in[14];
    const float* k_w   = (const float*)d_in[15];
    const float* k_b   = (const float*)d_in[16];
    const float* v_w   = (const float*)d_in[17];
    const float* v_b   = (const float*)d_in[18];
    const float* f_w   = (const float*)d_in[19];
    const float* f_b   = (const float*)d_in[20];

    const size_t MAP = (size_t)BB * CC * NN;          // 2M floats per [B,64,4096] map
    float* ws = (float*)d_ws;
    size_t off = 0;
    float* x1    = ws + off; off += MAP;
    float* img   = ws + off; off += MAP;
    float* tmp   = ws + off; off += MAP;
    float* ffeat = ws + off; off += MAP;
    float* bfeat = ws + off; off += MAP;
    float* vbuf  = ws + off; off += MAP;
    float* qbuf  = ws + off; off += (size_t)BB * QKC * NN;
    float* kbuf  = ws + off; off += (size_t)BB * QKC * NN;
    float* maxvb = ws + off; off += (size_t)BB * NN;
    int*   idxb  = (int*)(ws + off); off += (size_t)BB * NN;
    if (ws_size < off * sizeof(float)) return;

    const dim3 blk(256);
    const dim3 gLin(NN / 128, (BB * CC) / 64);        // (32, 8)
    const dim3 gConv(NN / 128, BB);                   // (32, 8)
    const dim3 gAmx(NN / 256, BB);                    // (16, 8)

    // shape_transform: two Linear+ReLU over the flattened spatial dim
    linear_relu_wmma<<<gLin, blk, 0, stream>>>(img_feat, st_w1, st_b1, x1,
                                               BB * CC, NN, 1024);
    linear_relu_wmma<<<gLin, blk, 0, stream>>>(x1, st_w2, st_b2, img,
                                               BB * CC, NN, NN);

    // ft: conv3x3+relu -> conv1x1+relu
    conv_wmma<3, 64, 0><<<gConv, blk, 0, stream>>>(img, nullptr, ft_w1, ft_b1, tmp,
                                                   nullptr, nullptr, nullptr, CC);
    conv_wmma<1, 64, 0><<<gConv, blk, 0, stream>>>(tmp, nullptr, ft_w2, ft_b2, ffeat,
                                                   nullptr, nullptr, nullptr, CC);
    // bt: conv3x3+relu -> conv1x1+relu
    conv_wmma<3, 64, 0><<<gConv, blk, 0, stream>>>(ffeat, nullptr, bt_w1, bt_b1, tmp,
                                                   nullptr, nullptr, nullptr, CC);
    conv_wmma<1, 64, 0><<<gConv, blk, 0, stream>>>(tmp, nullptr, bt_w2, bt_b2, bfeat,
                                                   nullptr, nullptr, nullptr, CC);

    // q, k, v 1x1 projections (no activation)
    conv_wmma<1, 64, 1><<<gConv, blk, 0, stream>>>(ffeat, nullptr, q_w, q_b, qbuf,
                                                   nullptr, nullptr, nullptr, QKC);
    conv_wmma<1, 64, 1><<<gConv, blk, 0, stream>>>(img, nullptr, k_w, k_b, kbuf,
                                                   nullptr, nullptr, nullptr, QKC);
    conv_wmma<1, 64, 1><<<gConv, blk, 0, stream>>>(bfeat, nullptr, v_w, v_b, vbuf,
                                                   nullptr, nullptr, nullptr, CC);

    // streaming hard-argmax over the N x N score matrix
    argmax_rows<<<gAmx, blk, 0, stream>>>(kbuf, qbuf, idxb, maxvb);

    // fusion conv3x3 over concat(f_feat, gather(v, idx)) with fused epilogue:
    //   out = f_feat + (conv + bias) * max_value
    conv_wmma<3, 128, 2><<<gConv, blk, 0, stream>>>(ffeat, vbuf, f_w, f_b, (float*)d_out,
                                                    idxb, maxvb, ffeat, CC);
}
